// WindowAttention_2499670966465
// MI455X (gfx1250) — matrix operand order repaired and checked by reading, not compiled, not run on hardware
//
#include <hip/hip_runtime.h>
#include <hip/hip_bf16.h>

// ---------- types ----------
typedef __bf16 bf16_t;
typedef __attribute__((ext_vector_type(8)))  bf16_t v8bf;
typedef __attribute__((ext_vector_type(16))) bf16_t v16bf;
typedef __attribute__((ext_vector_type(8)))  float  v8f;

union Frag { v16bf v; v8bf h[2]; bf16_t e[16]; };

// ---------- constants ----------
#define WIN_N   64
#define CDIM    512
#define HEADS   16
#define HD      32
#define QSTRIDE 1544                      // 1536 + 8 (bank-conflict pad)
#define ASTRIDE 520                       // 512 + 8
#define LDS_ATT_OFF  (64 * QSTRIDE)                  // bf16 elements
#define LDS_P_OFF    (LDS_ATT_OFF + 64 * ASTRIDE)
#define LDS_BF_ELEMS (LDS_P_OFF + 8 * 1024)
#define LDS_NRM_OFF  (LDS_BF_ELEMS * 2)              // bytes
#define LDS_BYTES    (LDS_NRM_OFF + 2048 * 4)        // = 288768 B (< 320KB/WGP)

// ---------- scalar conversions ----------
__device__ __forceinline__ bf16_t f2bf(float f) {
  unsigned u = __builtin_bit_cast(unsigned, f);
  u += 0x7FFFu + ((u >> 16) & 1u);                    // RNE
  unsigned short s = (unsigned short)(u >> 16);
  return __builtin_bit_cast(bf16_t, s);
}
__device__ __forceinline__ float bf2f(bf16_t b) {
  unsigned u = ((unsigned)__builtin_bit_cast(unsigned short, b)) << 16;
  return __builtin_bit_cast(float, u);
}

// ---------- fragment loaders (wave32 16-bit WMMA layouts) ----------
// A 16x32: lane holds row (lane&15); K = half*16+[0..16) contiguous in source row
__device__ __forceinline__ v16bf fragA(const bf16_t* rowp, int k0, int half) {
  Frag f;
  const bf16_t* p = rowp + k0 + half * 16;
  f.h[0] = *(const v8bf*)p;
  f.h[1] = *(const v8bf*)(p + 8);
  return f.v;
}
// B 32x16: lane holds col (lane&15); K = half*16+[0..16) contiguous in source row
__device__ __forceinline__ v16bf fragB(const bf16_t* rowp, int k0, int half) {
  Frag f;
  const bf16_t* p = rowp + k0 + half * 16;
  f.h[0] = *(const v8bf*)p;
  f.h[1] = *(const v8bf*)(p + 8);
  return f.v;
}

#define WMMA_BF16(A, B, C) \
  __builtin_amdgcn_wmma_f32_16x16x32_bf16(false, (A), false, (B), (short)0, (C), false, false)

// ---------- prep: fp32 weights -> bf16 workspace ----------
__global__ void prep_weights(const float* __restrict__ qkv_w,
                             const float* __restrict__ proj_w,
                             bf16_t* __restrict__ wq, bf16_t* __restrict__ wp) {
  const int nq = 1536 * 512, np = 512 * 512;
  for (int t = blockIdx.x * blockDim.x + threadIdx.x; t < nq + np;
       t += gridDim.x * blockDim.x) {
    if (t < nq) wq[t] = f2bf(qkv_w[t]);
    else        wp[t - nq] = f2bf(proj_w[t - nq]);
  }
}

// ---------- CPB MLP over the 15x15 log-spaced coord table ----------
__global__ void cpb_table(const float* __restrict__ w1, const float* __restrict__ b1,
                          const float* __restrict__ w2, float* __restrict__ bias_tab) {
  int p = blockIdx.x * blockDim.x + threadIdx.x;
  if (p >= 225) return;
  int i = p / 15, j = p % 15;
  float t0 = (float)(i - 7) * (8.0f / 7.0f);
  float t1 = (float)(j - 7) * (8.0f / 7.0f);
  t0 = (t0 >= 0.f ? 1.f : -1.f) * log2f(fabsf(t0) + 1.f) * (1.f / 3.f);
  t1 = (t1 >= 0.f ? 1.f : -1.f) * log2f(fabsf(t1) + 1.f) * (1.f / 3.f);
  float acc[HEADS];
#pragma unroll
  for (int h = 0; h < HEADS; ++h) acc[h] = 0.f;
  for (int k = 0; k < 512; ++k) {
    float hv = fmaxf(w1[2 * k] * t0 + w1[2 * k + 1] * t1 + b1[k], 0.f);
#pragma unroll
    for (int h = 0; h < HEADS; ++h) acc[h] += hv * w2[h * 512 + k];
  }
#pragma unroll
  for (int h = 0; h < HEADS; ++h) bias_tab[p * HEADS + h] = acc[h];
}

// ---------- expand to rpb[h][m][n] = 16*sigmoid(tab[idx(m,n)][h]) ----------
__global__ void cpb_rpb(const float* __restrict__ bias_tab, float* __restrict__ rpb) {
  int t = blockIdx.x * blockDim.x + threadIdx.x;   // 4096 = 64*64
  if (t >= WIN_N * WIN_N) return;
  int m = t >> 6, n = t & 63;
  int r0 = (m >> 3) - (n >> 3) + 7;
  int r1 = (m & 7) - (n & 7) + 7;
  int idx = r0 * 15 + r1;
#pragma unroll
  for (int h = 0; h < HEADS; ++h) {
    float v = bias_tab[idx * HEADS + h];
    rpb[((h << 6) + m) * WIN_N + n] = 16.f / (1.f + __expf(-v));
  }
}

// ---------- main fused window-attention kernel (1 workgroup / window) ----------
__global__ __launch_bounds__(256, 1)
void win_attn(const float* __restrict__ x,
              const bf16_t* __restrict__ wq, const bf16_t* __restrict__ wp,
              const float* __restrict__ q_bias, const float* __restrict__ v_bias,
              const float* __restrict__ flex, const float* __restrict__ rpb,
              const float* __restrict__ proj_b, float* __restrict__ out) {
  extern __shared__ char smem[];
  bf16_t* s_qkv = (bf16_t*)smem;                    // 64 x QSTRIDE (q|k|v)
  bf16_t* s_att = (bf16_t*)smem + LDS_ATT_OFF;      // 64 x ASTRIDE (x stage, then attn-out)
  bf16_t* s_p   = (bf16_t*)smem + LDS_P_OFF;        // 8 waves x (16 x 64) P strip
  float*  s_nrm = (float*)(smem + LDS_NRM_OFF);     // [2][16][64] inv-norms

  const int tid  = threadIdx.x;
  const int lane = tid & 31;
  const int wv   = tid >> 5;        // wave 0..7
  const int lm   = lane & 15;
  const int half = lane >> 4;
  const int b    = blockIdx.x;

  const float* xg = x + (size_t)b * (WIN_N * CDIM);

  // ---- stage x fp32 -> bf16 LDS ----
  for (int i = tid; i < WIN_N * CDIM / 4; i += 256) {
    int r = i >> 7;
    int c = (i & 127) << 2;
    const float4 v = *(const float4*)(xg + r * CDIM + c);
    bf16_t* d = s_att + r * ASTRIDE + c;
    d[0] = f2bf(v.x); d[1] = f2bf(v.y); d[2] = f2bf(v.z); d[3] = f2bf(v.w);
  }
  __syncthreads();

  // ---- Phase A: QKV = x @ Wqkv^T + bias (bf16 WMMA, fp32 acc) ----
  for (int j = 0; j < 12; ++j) {
    int ct = wv * 12 + j;                     // 96 column tiles of 16
    int n  = ct * 16 + lm;                    // 0..1535
    v8f acc[4] = {};
    const bf16_t* wrow = wq + (size_t)n * CDIM;
    for (int kk = 0; kk < 16; ++kk) {
      v16bf bfr = fragB(wrow, kk * 32, half);
#pragma unroll
      for (int rt = 0; rt < 4; ++rt) {
        v16bf af = fragA(s_att + (rt * 16 + lm) * ASTRIDE, kk * 32, half);
        acc[rt] = WMMA_BF16(af, bfr, acc[rt]);
      }
    }
    float bias = (n < 512) ? q_bias[n] : (n < 1024 ? 0.f : v_bias[n - 1024]);
#pragma unroll
    for (int rt = 0; rt < 4; ++rt)
#pragma unroll
      for (int i = 0; i < 8; ++i)
        s_qkv[(rt * 16 + i + half * 8) * QSTRIDE + n] = f2bf(acc[rt][i] + bias);
  }
  __syncthreads();

  // ---- inverse row norms for q and k (cosine attention) ----
  for (int t = tid; t < 2048; t += 256) {
    int isK = t >> 10, h = (t >> 6) & 15, m = t & 63;
    const bf16_t* p = s_qkv + m * QSTRIDE + isK * 512 + h * HD;
    float s = 0.f;
#pragma unroll
    for (int d = 0; d < HD; ++d) { float v = bf2f(p[d]); s += v * v; }
    s_nrm[t] = 1.f / fmaxf(sqrtf(s), 1e-12f);
  }
  __syncthreads();

  // ---- Phase B: per-head attention (2 heads per wave) ----
  bf16_t* Pw = s_p + wv * 1024;               // private 16x64 strip
  for (int hh = 0; hh < 2; ++hh) {
    const int h = wv * 2 + hh;
    const float scale_h = __expf(fminf(flex[h], 4.6051702f));  // ln(100)

    // V as B-operand (column gather from row-major LDS), hoisted for all strips
    v16bf vb[2][2];
#pragma unroll
    for (int kt = 0; kt < 2; ++kt)
#pragma unroll
      for (int cd = 0; cd < 2; ++cd) {
        Frag f;
        const int col = 1024 + h * HD + cd * 16 + lm;
#pragma unroll
        for (int e = 0; e < 16; ++e)
          f.e[e] = s_qkv[(kt * 32 + half * 16 + e) * QSTRIDE + col];
        vb[kt][cd] = f.v;
      }

    for (int rt = 0; rt < 4; ++rt) {
      // S strip = qn @ kn^T  (K=32 -> one WMMA per tile)
      v16bf qa = fragA(s_qkv + (rt * 16 + lm) * QSTRIDE + h * HD, 0, half);
      v8f s4[4];
#pragma unroll
      for (int cn = 0; cn < 4; ++cn) {
        v16bf kb = fragB(s_qkv + (cn * 16 + lm) * QSTRIDE + 512 + h * HD, 0, half);
        v8f z = {};
        s4[cn] = WMMA_BF16(qa, kb, z);
      }
      // cosine scale + temperature + relative position bias
#pragma unroll
      for (int cn = 0; cn < 4; ++cn) {
        const int n = cn * 16 + lm;
        const float ik = s_nrm[1024 + h * 64 + n] * scale_h;
#pragma unroll
        for (int i = 0; i < 8; ++i) {
          const int m = rt * 16 + i + half * 8;
          s4[cn][i] = s4[cn][i] * s_nrm[h * 64 + m] * ik +
                      rpb[((h << 6) + m) * WIN_N + n];
        }
      }
      // row softmax: reduce across 4 tiles + 16 lanes of the half
#pragma unroll
      for (int i = 0; i < 8; ++i) {
        float mx = fmaxf(fmaxf(s4[0][i], s4[1][i]), fmaxf(s4[2][i], s4[3][i]));
#pragma unroll
        for (int d = 1; d < 16; d <<= 1) mx = fmaxf(mx, __shfl_xor(mx, d, 32));
        float e0 = __expf(s4[0][i] - mx), e1 = __expf(s4[1][i] - mx);
        float e2 = __expf(s4[2][i] - mx), e3 = __expf(s4[3][i] - mx);
        float sum = e0 + e1 + e2 + e3;
#pragma unroll
        for (int d = 1; d < 16; d <<= 1) sum += __shfl_xor(sum, d, 32);
        const float inv = 1.f / sum;
        bf16_t* pr = Pw + (i + half * 8) * 64;
        pr[lm]      = f2bf(e0 * inv);
        pr[16 + lm] = f2bf(e1 * inv);
        pr[32 + lm] = f2bf(e2 * inv);
        pr[48 + lm] = f2bf(e3 * inv);
      }
      // O strip = P @ V
      v8f o[2] = {};
#pragma unroll
      for (int kt = 0; kt < 2; ++kt) {
        v16bf pa = fragA(Pw + lm * 64, kt * 32, half);
#pragma unroll
        for (int cd = 0; cd < 2; ++cd) o[cd] = WMMA_BF16(pa, vb[kt][cd], o[cd]);
      }
#pragma unroll
      for (int cd = 0; cd < 2; ++cd)
#pragma unroll
        for (int i = 0; i < 8; ++i)
          s_att[(rt * 16 + i + half * 8) * ASTRIDE + h * HD + cd * 16 + lm] =
              f2bf(o[cd][i]);
    }
  }
  __syncthreads();

  // ---- Phase C: out = attn_out @ Wp^T + b (fp32 to global) ----
  float* og = out + (size_t)b * (WIN_N * CDIM);
  for (int j = 0; j < 4; ++j) {
    const int ct = wv * 4 + j;                 // 32 column tiles
    const int n  = ct * 16 + lm;
    v8f acc[4] = {};
    const bf16_t* wrow = wp + (size_t)n * CDIM;
    for (int kk = 0; kk < 16; ++kk) {
      v16bf bfr = fragB(wrow, kk * 32, half);
#pragma unroll
      for (int rt = 0; rt < 4; ++rt) {
        v16bf af = fragA(s_att + (rt * 16 + lm) * ASTRIDE, kk * 32, half);
        acc[rt] = WMMA_BF16(af, bfr, acc[rt]);
      }
    }
    const float pb = proj_b[n];
#pragma unroll
    for (int rt = 0; rt < 4; ++rt)
#pragma unroll
      for (int i = 0; i < 8; ++i)
        og[(rt * 16 + i + half * 8) * CDIM + n] = acc[rt][i] + pb;
  }
}

// ---------- launcher ----------
extern "C" void kernel_launch(void* const* d_in, const int* in_sizes, int n_in,
                              void* d_out, int out_size, void* d_ws, size_t ws_size,
                              hipStream_t stream) {
  const float* x      = (const float*)d_in[0];
  const float* qkv_w  = (const float*)d_in[1];
  const float* q_bias = (const float*)d_in[2];
  const float* v_bias = (const float*)d_in[3];
  const float* flex   = (const float*)d_in[4];
  const float* cpb_w1 = (const float*)d_in[5];
  const float* cpb_b1 = (const float*)d_in[6];
  const float* cpb_w2 = (const float*)d_in[7];
  const float* proj_w = (const float*)d_in[8];
  const float* proj_b = (const float*)d_in[9];
  float* out = (float*)d_out;

  char* ws = (char*)d_ws;                      // needs ~2.4 MB
  bf16_t* wq       = (bf16_t*)(ws);            // 1536*512*2 = 1,572,864 B
  bf16_t* wp       = (bf16_t*)(ws + 1572864);  //  512*512*2 =   524,288 B
  float*  bias_tab = (float*)(ws + 2097152);   //  225*16*4  =    14,400 B
  float*  rpb      = (float*)(ws + 2113536);   // 16*64*64*4 =   262,144 B

  prep_weights<<<512, 256, 0, stream>>>(qkv_w, proj_w, wq, wp);
  cpb_table<<<1, 256, 0, stream>>>(cpb_w1, cpb_b1, cpb_w2, bias_tab);
  cpb_rpb<<<16, 256, 0, stream>>>(bias_tab, rpb);
  win_attn<<<2048, 256, LDS_BYTES, stream>>>(x, wq, wp, q_bias, v_bias, flex,
                                             rpb, proj_b, out);
}
